// SpreadGNN_60876866453974
// MI455X (gfx1250) — matrix operand order repaired and checked by reading, not compiled, not run on hardware
//
#include <hip/hip_runtime.h>

// ---------------------------------------------------------------------------
// SpreadGNN forward for MI455X (gfx1250, wave32, WMMA).
//
// Fully-connected-with-self-loop 32-node graphs => SAGE aggregation is a
// per-graph column mean, identical for every node:
//   h' = relu( h @ Wr^T + broadcast(mean_rows(h) @ Wl^T) + b )
// One wave32 per graph; h lives in LDS (32 x F bf16, 272B row pitch for
// 16B-aligned, bank-spread b128 access). GEMMs -> v_wmma_f32_16x16x32_bf16,
// mean path folded into the accumulator via a broadcast-A fragment with the
// bias preloaded into C. Column means are produced for free at writeback
// (C-layout partial sums + one shfl_xor(16)) and stored as bf16, so both the
// A fragments AND the broadcast-mean fragments are pure ds_load_b128 pairs.
// Weights are repacked once into bf16 B-fragment-native layout in the
// workspace: the hot loop's B loads are two coalesced global_load_b128 per
// fragment (WGP$/L2-resident), prefetched one N-tile ahead.
// ---------------------------------------------------------------------------

typedef __bf16 bf16;
typedef __attribute__((ext_vector_type(16))) __bf16 v16bf;
typedef __attribute__((ext_vector_type(8)))  __bf16 v8bf;
typedef __attribute__((ext_vector_type(8)))  float  v8f;

#define STRIDE 136   // bf16 elems per LDS row: 272B pitch = 17*16B (aligned, bank-spread)

// ---- packed-weight fragment layout ----------------------------------------
// One fragment = 32 lanes x 16 bf16 = 1KB, indexed [nt*KT + kt][lane].
// Offsets in v16bf (32B) units within the workspace:
#define FR_W1L 0      //  4 frags (NT=4, KT=1)
#define FR_W1R 4      //  4
#define FR_W2L 8      // 16 frags (NT=8, KT=2)
#define FR_W2R 24     // 16
#define FR_W3L 40     // 32 frags (NT=8, KT=4)
#define FR_W3R 72     // 32
#define FR_WP  104    //  4 frags (NT=1, KT=4)
#define FR_TOTAL 108
#define PACK_BYTES (FR_TOTAL * 32 * 32)   // 110,592 B

// Combine two contiguous 8-element bf16 runs into a WMMA A-operand register set.
__device__ inline v16bf combine16(const v8bf lo, const v8bf hi) {
  v16bf a;
#pragma unroll
  for (int i = 0; i < 8; ++i) { a[i] = lo[i]; a[8 + i] = hi[i]; }
  return a;
}

// A-fragment (16-bit A, 16x32) from LDS: lane l -> row mtile*16+(l&15);
// per-lane K pattern = runs K=base+0..7 and K=base+16..23 -> two b128 loads.
__device__ inline v16bf a_frag_lds(const bf16* h, int mtile, int ktile, int lane) {
  const int row = mtile * 16 + (lane & 15);
  const int g   = (lane >> 4) & 1;
  const bf16* rp = h + row * STRIDE + ktile * 32 + g * 8;
  return combine16(*(const v8bf*)(rp), *(const v8bf*)(rp + 16));
}

// Broadcast-A fragment: every row = mean vector m (bf16, plain order in LDS).
// Same per-lane K pattern as a_frag -> two b128 loads, no converts.
__device__ inline v16bf m_frag(const bf16* m, int ktile, int lane) {
  const int g = (lane >> 4) & 1;
  const bf16* mp = m + ktile * 32 + g * 8;
  return combine16(*(const v8bf*)(mp), *(const v8bf*)(mp + 16));
}

// Direct B-fragment from row-major fp32 W[F_out][F_in] (pack kernel + fallback).
__device__ inline v16bf b_frag_direct(const float* __restrict__ W, int F_in, int F_out,
                                      int ntile, int ktile, int lane) {
  const int n  = ntile * 16 + (lane & 15);
  const int g  = (lane >> 4) & 1;
  const int kb = ktile * 32 + g * 8;
  const bool nok = (n < F_out);
  const float* wp = W + n * F_in + kb;
  v16bf b;
#pragma unroll
  for (int v = 0; v < 8; ++v) {
    const int o = 2 * v + (v >= 4 ? 8 : 0);
    const int k = kb + o;
    const float x0 = (nok && (k     < F_in)) ? wp[o]     : 0.f;
    const float x1 = (nok && (k + 1 < F_in)) ? wp[o + 1] : 0.f;
    b[2 * v]     = (bf16)x0;
    b[2 * v + 1] = (bf16)x1;
  }
  return b;
}

template <bool PACKED>
__device__ inline v16bf get_b(const float* __restrict__ W, const v16bf* __restrict__ pw,
                              int F_in, int F_out, int KT, int nt, int kt, int lane) {
  if (PACKED) return pw[(nt * KT + kt) * 32 + lane];   // two global_load_b128
  return b_frag_direct(W, F_in, F_out, nt, kt, lane);
}

// ---- one-time weight repack into B-fragment-native bf16 layout ------------
__global__ __launch_bounds__(32)
void pack_weights_kernel(const float* __restrict__ W, int F_in, int F_out, int KT,
                         v16bf* __restrict__ out) {
  const int nt = blockIdx.x / KT;
  const int kt = blockIdx.x % KT;
  out[blockIdx.x * 32 + threadIdx.x] =
      b_frag_direct(W, F_in, F_out, nt, kt, threadIdx.x);
}

// ---- one SAGE layer -------------------------------------------------------
// hout = relu( hin @ Wr^T + broadcast(m_in @ Wl^T) + b ); also emits the
// per-graph column mean of hout into m_out (bf16), fused into the writeback.
template <int KT, int NT, bool PACKED>
__device__ void sage_layer(const bf16* hin, bf16* hout,
                           const bf16* m_in, bf16* m_out,
                           const float* __restrict__ Wl, const float* __restrict__ Wr,
                           const v16bf* __restrict__ pWl, const v16bf* __restrict__ pWr,
                           const float* __restrict__ bias,
                           int F_in, int F_out, int lane) {
  const int ncol = lane & 15;
  const int rb   = 8 * (lane >> 4);
  for (int nt = 0; nt < NT; ++nt) {
    if (PACKED && nt + 1 < NT) {   // global_prefetch_b8, near-cache locality
      __builtin_prefetch(&pWl[((nt + 1) * KT) * 32 + lane], 0, 3);
      __builtin_prefetch(&pWr[((nt + 1) * KT) * 32 + lane], 0, 3);
    }
    const float bv = (nt * 16 + ncol < F_out) ? bias[nt * 16 + ncol] : 0.f;
    v8f cm, c0, c1;
#pragma unroll
    for (int v = 0; v < 8; ++v) { cm[v] = bv; c0[v] = 0.f; c1[v] = 0.f; }

#pragma unroll
    for (int kt = 0; kt < KT; ++kt) {
      const v16bf bl = get_b<PACKED>(Wl, pWl, F_in, F_out, KT, nt, kt, lane);
      const v16bf br = get_b<PACKED>(Wr, pWr, F_in, F_out, KT, nt, kt, lane);
      const v16bf am = m_frag(m_in, kt, lane);
      const v16bf a0 = a_frag_lds(hin, 0, kt, lane);
      const v16bf a1 = a_frag_lds(hin, 1, kt, lane);
      cm = __builtin_amdgcn_wmma_f32_16x16x32_bf16(false, am, false, bl, (short)0, cm, false, false);
      c0 = __builtin_amdgcn_wmma_f32_16x16x32_bf16(false, a0, false, br, (short)0, c0, false, false);
      c1 = __builtin_amdgcn_wmma_f32_16x16x32_bf16(false, a1, false, br, (short)0, c1, false, false);
    }

    // C/D layout: VGPR v, lane l -> row v + 8*(l>>4) (+16 for tile 1), col l&15.
    // Each lane holds 16 of the 32 rows of its column -> partial column sum;
    // lane l^16 holds the other 16 rows.
    float csum = 0.f;
#pragma unroll
    for (int v = 0; v < 8; ++v) {
      float y0 = fmaxf(c0[v] + cm[v], 0.f);
      float y1 = fmaxf(c1[v] + cm[v], 0.f);
      csum += y0 + y1;
      hout[(rb + v) * STRIDE + nt * 16 + ncol]      = (bf16)y0;
      hout[(16 + rb + v) * STRIDE + nt * 16 + ncol] = (bf16)y1;
    }
    csum += __shfl_xor(csum, 16, 32);
    if (lane < 16) m_out[nt * 16 + ncol] = (bf16)(csum * 0.03125f);
  }
  __syncthreads();
}

// ---- head: out = mean_rows(h3) @ Wp^T + bp (mean already in m_in) ---------
template <bool PACKED>
__device__ void head(const bf16* m_in,
                     const float* __restrict__ Wp, const v16bf* __restrict__ pWp,
                     const float* __restrict__ bp,
                     float* __restrict__ out, int g, int lane) {
  const int ncol = lane & 15;
  const float bv = (ncol < 5) ? bp[ncol] : 0.f;
  v8f cm;
#pragma unroll
  for (int v = 0; v < 8; ++v) cm[v] = bv;
#pragma unroll
  for (int kt = 0; kt < 4; ++kt) {
    const v16bf b  = get_b<PACKED>(Wp, pWp, 128, 5, 4, 0, kt, lane);
    const v16bf am = m_frag(m_in, kt, lane);
    cm = __builtin_amdgcn_wmma_f32_16x16x32_bf16(false, am, false, b, (short)0, cm, false, false);
  }
  // All rows identical; row 0 = VGPR 0, lanes 0..15 (col = lane).
  if (lane < 5) out[g * 5 + lane] = cm[0];
}

// ---- main kernel ----------------------------------------------------------
template <bool PACKED>
__global__ __launch_bounds__(32)
void spread_gnn_kernel(const float* __restrict__ obs,
                       const float* __restrict__ W1l, const float* __restrict__ b1,
                       const float* __restrict__ W1r,
                       const float* __restrict__ W2l, const float* __restrict__ b2,
                       const float* __restrict__ W2r,
                       const float* __restrict__ W3l, const float* __restrict__ b3,
                       const float* __restrict__ W3r,
                       const float* __restrict__ Wp,  const float* __restrict__ bp,
                       const v16bf* __restrict__ pws,
                       float* __restrict__ out) {
  __shared__ __align__(16) bf16 hA[32 * STRIDE];
  __shared__ __align__(16) bf16 hB[32 * STRIDE];
  __shared__ __align__(16) bf16 mA[128];
  __shared__ __align__(16) bf16 mB[128];

  const int g    = blockIdx.x;   // one graph per wave32 block
  const int lane = threadIdx.x;  // lane == node row

  // Load x[32 x 5] (K zero-padded to 32) and butterfly-reduce the column mean.
  {
    const float* xr = obs + g * 160 + lane * 5;
    float xf[5];
#pragma unroll
    for (int f = 0; f < 5; ++f) {
      xf[f] = xr[f];
      hA[lane * STRIDE + f] = (bf16)xf[f];
    }
#pragma unroll
    for (int f = 5; f < 32; ++f) hA[lane * STRIDE + f] = (bf16)0.f;
#pragma unroll
    for (int off = 16; off >= 1; off >>= 1)
#pragma unroll
      for (int f = 0; f < 5; ++f) xf[f] += __shfl_xor(xf[f], off, 32);
    if (lane == 0) {
#pragma unroll
      for (int f = 0; f < 5; ++f)  mA[f] = (bf16)(xf[f] * 0.03125f);
#pragma unroll
      for (int f = 5; f < 32; ++f) mA[f] = (bf16)0.f;
    }
  }
  __syncthreads();

  sage_layer<1, 4, PACKED>(hA, hB, mA, mB, W1l, W1r, pws + FR_W1L, pws + FR_W1R,
                           b1, 5, 64, lane);
  sage_layer<2, 8, PACKED>(hB, hA, mB, mA, W2l, W2r, pws + FR_W2L, pws + FR_W2R,
                           b2, 64, 128, lane);
  sage_layer<4, 8, PACKED>(hA, hB, mA, mB, W3l, W3r, pws + FR_W3L, pws + FR_W3R,
                           b3, 128, 128, lane);
  head<PACKED>(mB, Wp, pws + FR_WP, bp, out, g, lane);
}

extern "C" void kernel_launch(void* const* d_in, const int* in_sizes, int n_in,
                              void* d_out, int out_size, void* d_ws, size_t ws_size,
                              hipStream_t stream) {
  (void)in_sizes; (void)n_in; (void)out_size;
  const float* obs = (const float*)d_in[0];
  // d_in[1]=src, d_in[2]=dst unused: fully-connected + self-loop topology
  // collapses aggregation into a per-graph mean.
  const float* W1l = (const float*)d_in[3];
  const float* b1  = (const float*)d_in[4];
  const float* W1r = (const float*)d_in[5];
  const float* W2l = (const float*)d_in[6];
  const float* b2  = (const float*)d_in[7];
  const float* W2r = (const float*)d_in[8];
  const float* W3l = (const float*)d_in[9];
  const float* b3  = (const float*)d_in[10];
  const float* W3r = (const float*)d_in[11];
  const float* Wp  = (const float*)d_in[12];
  const float* bp  = (const float*)d_in[13];
  float* out = (float*)d_out;
  v16bf* pws = (v16bf*)d_ws;

  if (ws_size >= (size_t)PACK_BYTES && pws != nullptr) {
    // One-time (per launch) repack of all weights into bf16 fragment layout.
    pack_weights_kernel<<< 4, 32, 0, stream>>>(W1l,   5,  64, 1, pws + FR_W1L);
    pack_weights_kernel<<< 4, 32, 0, stream>>>(W1r,   5,  64, 1, pws + FR_W1R);
    pack_weights_kernel<<<16, 32, 0, stream>>>(W2l,  64, 128, 2, pws + FR_W2L);
    pack_weights_kernel<<<16, 32, 0, stream>>>(W2r,  64, 128, 2, pws + FR_W2R);
    pack_weights_kernel<<<32, 32, 0, stream>>>(W3l, 128, 128, 4, pws + FR_W3L);
    pack_weights_kernel<<<32, 32, 0, stream>>>(W3r, 128, 128, 4, pws + FR_W3R);
    pack_weights_kernel<<< 4, 32, 0, stream>>>(Wp,  128,   5, 4, pws + FR_WP);
    spread_gnn_kernel<true><<<2048, 32, 0, stream>>>(
        obs, W1l, b1, W1r, W2l, b2, W2r, W3l, b3, W3r, Wp, bp, pws, out);
  } else {
    // Fallback: build B fragments straight from fp32 weights.
    spread_gnn_kernel<false><<<2048, 32, 0, stream>>>(
        obs, W1l, b1, W1r, W2l, b2, W2r, W3l, b3, W3r, Wp, bp, pws, out);
  }
}
